// LearnableHash_49907519979535
// MI455X (gfx1250) — matrix operand order repaired and checked by reading, not compiled, not run on hardware
//
#include <hip/hip_runtime.h>
#include <hip/hip_bf16.h>

typedef __attribute__((ext_vector_type(16))) _Float16 v16h;
typedef __attribute__((ext_vector_type(8)))  float    v8f;

#define RADIUS_F 1.3f
#define NI 128
#define STEP_F (2.0f * RADIUS_F / 128.0f)

// transposed f16 weight layout in workspace (element offsets)
#define W0T_OFF 0       // [128][32]
#define W1T_OFF 4096    // [128][128]
#define W2T_OFF 20480   // [16][128]
#define C0T_OFF 22528   // [64][32]  (k=31 zero-padded)
#define C1T_OFF 24576   // [64][64]
#define C2T_OFF 28672   // [16][64]  (n>=3 zero-padded)
#define WT_TOTAL 29696

__global__ void prep_weights(const float* __restrict__ s_w0, const float* __restrict__ s_w1,
                             const float* __restrict__ s_w2, const float* __restrict__ c_w0,
                             const float* __restrict__ c_w1, const float* __restrict__ c_w2,
                             _Float16* __restrict__ wt) {
  int idx = blockIdx.x * blockDim.x + threadIdx.x;
  if (idx < 128 * 32)  { int n = idx >> 5, k = idx & 31;  wt[W0T_OFF + idx] = (_Float16)s_w0[k * 128 + n]; }
  if (idx < 128 * 128) { int n = idx >> 7, k = idx & 127; wt[W1T_OFF + idx] = (_Float16)s_w1[k * 128 + n]; }
  if (idx < 16 * 128)  { int n = idx >> 7, k = idx & 127; wt[W2T_OFF + idx] = (_Float16)s_w2[k * 16 + n]; }
  if (idx < 64 * 32)   { int n = idx >> 5, k = idx & 31;  wt[C0T_OFF + idx] = (_Float16)((k < 31) ? c_w0[k * 64 + n] : 0.f); }
  if (idx < 64 * 64)   { int n = idx >> 6, k = idx & 63;  wt[C1T_OFF + idx] = (_Float16)c_w1[k * 64 + n]; }
  if (idx < 16 * 64)   { int n = idx >> 6, k = idx & 63;  wt[C2T_OFF + idx] = (_Float16)((n < 3) ? c_w2[k * 3 + n] : 0.f); }
}

__device__ __forceinline__ void lds_fence() {
#if __has_builtin(__builtin_amdgcn_wave_barrier)
  __builtin_amdgcn_wave_barrier();
#endif
}

__device__ __forceinline__ v8f wmma16(v16h a, v16h b, v8f c) {
  return __builtin_amdgcn_wmma_f32_16x16x32_f16(false, a, false, b, (short)0, c, false, false);
}

// A fragment (16xK tile, f16, row-major ld): lane<16 -> K {0..7,16..23}, lane>=16 -> K {8..15,24..31}
__device__ __forceinline__ v16h load_a_frag(const _Float16* base, int ld, int lane) {
  int m = lane & 15;
  int kb = (lane < 16) ? 0 : 8;
  const _Float16* p = base + m * ld + kb;
  union { v16h v; uint4 q[2]; } r;
  r.q[0] = *(const uint4*)(p);
  r.q[1] = *(const uint4*)(p + 16);
  return r.v;
}

// B fragment from transposed weights wt[N][K]: lane<16 -> K {0..7,16..23}, lane>=16 -> K {8..15,24..31} of column n
__device__ __forceinline__ v16h load_b_frag(const _Float16* wt, int ld, int ntile, int kchunk, int lane) {
  int n = ntile * 16 + (lane & 15);
  int kb = kchunk * 32 + ((lane < 16) ? 0 : 8);
  const _Float16* p = wt + n * ld + kb;
  union { v16h v; uint4 q[2]; } r;
  r.q[0] = *(const uint4*)(p);
  r.q[1] = *(const uint4*)(p + 16);
  return r.v;
}

// D fragment (16x16 f32): lane -> n, vgpr r -> row r + (lane<16?0:8)
__device__ __forceinline__ void store_d_relu(v8f d, _Float16* outp, int ld, int n0, int lane) {
  int n = n0 + (lane & 15);
  int mb = (lane < 16) ? 0 : 8;
#pragma unroll
  for (int r = 0; r < 8; ++r) {
    float v = d[r];
    outp[(mb + r) * ld + n] = (_Float16)(v > 0.f ? v : 0.f);
  }
}

__device__ __forceinline__ void store_d_f32(v8f d, float* buf, int lane) {
  int n = lane & 15;
  int mb = (lane < 16) ? 0 : 8;
#pragma unroll
  for (int r = 0; r < 8; ++r) buf[(mb + r) * 16 + n] = d[r];
}

__device__ __forceinline__ void tricoord(float c, int dim, int& i0, int& i1, float& w) {
  float f = (c + 1.f) * 0.5f * (float)(dim - 1);
  f = fminf(fmaxf(f, 0.f), (float)(dim - 1));
  float fl = floorf(f);
  i0 = (int)fl;
  i1 = min(i0 + 1, dim - 1);
  w = f - fl;
}

__global__ __launch_bounds__(128) void nerf_main(
    const float* __restrict__ rays_o, const float* __restrict__ rays_d,
    const float* __restrict__ G1, const float* __restrict__ Fg,
    const _Float16* __restrict__ wt, float* __restrict__ out) {
  __shared__ _Float16 s_wt[WT_TOTAL];
  __shared__ _Float16 s_act[4][2][16 * 128];
  __shared__ float    s_sig[4][256];
  __shared__ float    s_rgb[4][256];

  const int tid = threadIdx.x;
  const int wave = tid >> 5;
  const int lane = tid & 31;

  // stage f16 weights into LDS (uint4 = 8 halves per load)
  for (int i = tid * 8; i < WT_TOTAL; i += 128 * 8)
    *(uint4*)&s_wt[i] = *(const uint4*)&wt[i];
  __syncthreads();

  const _Float16* w0t = s_wt + W0T_OFF;
  const _Float16* w1t = s_wt + W1T_OFF;
  const _Float16* w2t = s_wt + W2T_OFF;
  const _Float16* c0t = s_wt + C0T_OFF;
  const _Float16* c1t = s_wt + C1T_OFF;
  const _Float16* c2t = s_wt + C2T_OFF;

  _Float16* actA = s_act[wave][0];
  _Float16* actB = s_act[wave][1];
  float*    sbuf = s_sig[wave];
  float*    rbuf = s_rgb[wave];

  const int ray = blockIdx.x * 4 + wave;
  const float ox = rays_o[ray * 3 + 0], oy = rays_o[ray * 3 + 1], oz = rays_o[ray * 3 + 2];
  const float dxr = rays_d[ray * 3 + 0], dyr = rays_d[ray * 3 + 1], dzr = rays_d[ray * 3 + 2];

  // ray-AABB
  float sdx = (fabsf(dxr) < 1e-9f) ? 1e-9f : dxr;
  float sdy = (fabsf(dyr) < 1e-9f) ? 1e-9f : dyr;
  float sdz = (fabsf(dzr) < 1e-9f) ? 1e-9f : dzr;
  float t1x = (-RADIUS_F - ox) / sdx, t2x = (RADIUS_F - ox) / sdx;
  float t1y = (-RADIUS_F - oy) / sdy, t2y = (RADIUS_F - oy) / sdy;
  float t1z = (-RADIUS_F - oz) / sdz, t2z = (RADIUS_F - oz) / sdz;
  float tmin = fmaxf(fmaxf(fminf(t1x, t2x), fminf(t1y, t2y)), fminf(t1z, t2z));
  float tmax = fminf(fminf(fmaxf(t1x, t2x), fmaxf(t1y, t2y)), fmaxf(t1z, t2z));
  float tnear = fmaxf(tmin, 0.f);

  // SH deg-4 encoding of normalized direction (uniform per wave)
  float ilen = rsqrtf(dxr * dxr + dyr * dyr + dzr * dzr);
  float nx = dxr * ilen, ny = dyr * ilen, nz = dzr * ilen;
  float x2 = nx * nx, y2 = ny * ny, z2 = nz * nz;
  float xy = nx * ny, yz = ny * nz, xz = nx * nz;
  float enc[16] = {
      0.28209479177387814f,
      -0.48860251190291987f * ny,
      0.48860251190291987f * nz,
      -0.48860251190291987f * nx,
      1.0925484305920792f * xy,
      -1.0925484305920792f * yz,
      0.94617469575756f * z2 - 0.31539156525252f,
      -1.0925484305920792f * xz,
      0.5462742152960396f * (x2 - y2),
      0.5900435899266435f * ny * (-3.0f * x2 + y2),
      2.890611442640554f * xy * nz,
      0.4570457994644657f * ny * (1.0f - 5.0f * z2),
      0.3731763325901154f * nz * (5.0f * z2 - 3.0f),
      0.4570457994644657f * nx * (1.0f - 5.0f * z2),
      1.445305721320277f * nz * (x2 - y2),
      0.5900435899266435f * nx * (-x2 + 3.0f * y2)};

  const int m = lane & 15;   // sample row within tile
  const int h = lane >> 4;   // z-corner half for Fg sampling

  float T = 1.f;
  float accR = 0.f, accG = 0.f, accB = 0.f, accW = 0.f;

#pragma unroll 1
  for (int tile = 0; tile < 8; ++tile) {
    float t = tnear + (float)(tile * 16 + m) * STEP_F;
    float cx = (ox + dxr * t) * (1.f / RADIUS_F);
    float cy = (oy + dyr * t) * (1.f / RADIUS_F);
    float cz = (oz + dzr * t) * (1.f / RADIUS_F);

    // ---- G1 trilinear (3 channels) ----
    int gx0, gx1, gy0, gy1, gz0, gz1;
    float gwx, gwy, gwz;
    tricoord(cx, 256, gx0, gx1, gwx);
    tricoord(cy, 256, gy0, gy1, gwy);
    tricoord(cz, 256, gz0, gz1, gwz);
    float g0 = 0.f, g1v = 0.f, g2 = 0.f;
#pragma unroll
    for (int dz = 0; dz < 2; ++dz) {
      int iz = dz ? gz1 : gz0;
      float wwz = dz ? gwz : 1.f - gwz;
#pragma unroll
      for (int dy = 0; dy < 2; ++dy) {
        int iy = dy ? gy1 : gy0;
        float wwy = dy ? gwy : 1.f - gwy;
#pragma unroll
        for (int dxc = 0; dxc < 2; ++dxc) {
          int ix = dxc ? gx1 : gx0;
          float w = wwz * wwy * (dxc ? gwx : 1.f - gwx);
          const float* p = G1 + (size_t)((iz * 256 + iy) * 256 + ix) * 3;
          g0 += w * p[0]; g1v += w * p[1]; g2 += w * p[2];
        }
      }
    }

    // ---- Fg trilinear (32 channels), z-corner split across half-waves ----
    int fx0, fx1, fy0, fy1, fz0, fz1;
    float fwx, fwy, fwz;
    tricoord(g0, 64, fx0, fx1, fwx);
    tricoord(g1v, 64, fy0, fy1, fwy);
    tricoord(g2, 64, fz0, fz1, fwz);
    float fv[32];
#pragma unroll
    for (int c = 0; c < 32; ++c) fv[c] = 0.f;
    {
      int iz = h ? fz1 : fz0;
      float wwz = h ? fwz : 1.f - fwz;
#pragma unroll
      for (int dy = 0; dy < 2; ++dy) {
        int iy = dy ? fy1 : fy0;
        float wwy = dy ? fwy : 1.f - fwy;
#pragma unroll
        for (int dxc = 0; dxc < 2; ++dxc) {
          int ix = dxc ? fx1 : fx0;
          float w = wwz * wwy * (dxc ? fwx : 1.f - fwx);
          const float4* p = (const float4*)(Fg + (size_t)((iz * 64 + iy) * 64 + ix) * 32);
#pragma unroll
          for (int j = 0; j < 8; ++j) {
            float4 v = p[j];
            fv[4 * j + 0] += w * v.x;
            fv[4 * j + 1] += w * v.y;
            fv[4 * j + 2] += w * v.z;
            fv[4 * j + 3] += w * v.w;
          }
        }
      }
    }
#pragma unroll
    for (int c = 0; c < 32; ++c) fv[c] += __shfl_xor(fv[c], 16);

    // store X tile (16x32 f16) into actA, ld=128
    if (lane < 16) {
      uint4* row = (uint4*)&actA[m * 128];
#pragma unroll
      for (int q = 0; q < 4; ++q) {
        union { _Float16 hh[8]; uint4 v; } u;
#pragma unroll
        for (int e = 0; e < 8; ++e) u.hh[e] = (_Float16)fv[q * 8 + e];
        row[q] = u.v;
      }
    }
    lds_fence();

    // ---- sigma net layer 0: [16x32] x [32x128] ----
    {
      v16h ax = load_a_frag(actA, 128, lane);
#pragma unroll
      for (int nt = 0; nt < 8; ++nt) {
        v8f acc = {0.f, 0.f, 0.f, 0.f, 0.f, 0.f, 0.f, 0.f};
        acc = wmma16(ax, load_b_frag(w0t, 32, nt, 0, lane), acc);
        store_d_relu(acc, actB, 128, nt * 16, lane);
      }
    }
    lds_fence();
    // ---- sigma net layer 1: [16x128] x [128x128] ----
    {
      v16h a1[4];
#pragma unroll
      for (int kc = 0; kc < 4; ++kc) a1[kc] = load_a_frag(actB + kc * 32, 128, lane);
#pragma unroll
      for (int nt = 0; nt < 8; ++nt) {
        v8f acc = {0.f, 0.f, 0.f, 0.f, 0.f, 0.f, 0.f, 0.f};
#pragma unroll
        for (int kc = 0; kc < 4; ++kc) acc = wmma16(a1[kc], load_b_frag(w1t, 128, nt, kc, lane), acc);
        store_d_relu(acc, actA, 128, nt * 16, lane);
      }
    }
    lds_fence();
    // ---- sigma net layer 2: [16x128] x [128x16] -> S (f32) ----
    {
      v8f acc = {0.f, 0.f, 0.f, 0.f, 0.f, 0.f, 0.f, 0.f};
#pragma unroll
      for (int kc = 0; kc < 4; ++kc)
        acc = wmma16(load_a_frag(actA + kc * 32, 128, lane), load_b_frag(w2t, 128, 0, kc, lane), acc);
      store_d_f32(acc, sbuf, lane);
    }
    lds_fence();

    // ---- build cin tile [16x32]: cols 0..15 = enc, 16..30 = sig_out[1..15], 31 = 0 ----
    if (lane < 16) {
      _Float16* row = actA + m * 128;
#pragma unroll
      for (int j = 0; j < 16; ++j) row[j] = (_Float16)enc[j];
#pragma unroll
      for (int j = 1; j < 16; ++j) row[15 + j] = (_Float16)sbuf[m * 16 + j];
      row[31] = (_Float16)0.f;
    }
    lds_fence();

    // ---- color layer 0: [16x32] x [32x64] ----
    {
      v16h ac = load_a_frag(actA, 128, lane);
#pragma unroll
      for (int nt = 0; nt < 4; ++nt) {
        v8f acc = {0.f, 0.f, 0.f, 0.f, 0.f, 0.f, 0.f, 0.f};
        acc = wmma16(ac, load_b_frag(c0t, 32, nt, 0, lane), acc);
        store_d_relu(acc, actB, 128, nt * 16, lane);
      }
    }
    lds_fence();
    // ---- color layer 1: [16x64] x [64x64] ----
    {
      v16h a1c[2];
#pragma unroll
      for (int kc = 0; kc < 2; ++kc) a1c[kc] = load_a_frag(actB + kc * 32, 128, lane);
#pragma unroll
      for (int nt = 0; nt < 4; ++nt) {
        v8f acc = {0.f, 0.f, 0.f, 0.f, 0.f, 0.f, 0.f, 0.f};
#pragma unroll
        for (int kc = 0; kc < 2; ++kc) acc = wmma16(a1c[kc], load_b_frag(c1t, 64, nt, kc, lane), acc);
        store_d_relu(acc, actA, 128, nt * 16, lane);
      }
    }
    lds_fence();
    // ---- color layer 2: [16x64] x [64x16] -> rgb (f32, cols 0..2 valid) ----
    {
      v8f acc = {0.f, 0.f, 0.f, 0.f, 0.f, 0.f, 0.f, 0.f};
#pragma unroll
      for (int kc = 0; kc < 2; ++kc)
        acc = wmma16(load_a_frag(actA + kc * 32, 128, lane), load_b_frag(c2t, 64, 0, kc, lane), acc);
      store_d_f32(acc, rbuf, lane);
    }
    lds_fence();

    // ---- volume rendering for this 16-sample tile ----
    {
      float sig0 = sbuf[m * 16 + 0];
      bool valid = (t < tmax) && (tmax > tnear);
      float sigma = valid ? fmaxf(sig0, 0.f) : 0.f;
      float alpha = 1.f - expf(-sigma * STEP_F);
      float q = 1.f - alpha + 1e-10f;
      float p = q;
#pragma unroll
      for (int off = 1; off < 16; off <<= 1) {
        float u = __shfl_up(p, off, 16);
        if (m >= off) p *= u;
      }
      float excl = __shfl_up(p, 1, 16);
      if (m == 0) excl = 1.f;
      float w = alpha * (T * excl);
      float rr = rbuf[m * 16 + 0];
      float rg = rbuf[m * 16 + 1];
      float rb = rbuf[m * 16 + 2];
      accR += w * (1.f / (1.f + expf(-rr)));
      accG += w * (1.f / (1.f + expf(-rg)));
      accB += w * (1.f / (1.f + expf(-rb)));
      accW += w;
      T *= __shfl(p, 15, 16);
    }
    lds_fence();
  }

  // reduce across the 16 samples (lanes 16..31 hold duplicates; width-16 keeps them separate)
#pragma unroll
  for (int off = 8; off > 0; off >>= 1) {
    accR += __shfl_down(accR, off, 16);
    accG += __shfl_down(accG, off, 16);
    accB += __shfl_down(accB, off, 16);
    accW += __shfl_down(accW, off, 16);
  }
  if (lane == 0) {
    float bg = 1.f - accW;
    out[ray * 3 + 0] = accR + bg;
    out[ray * 3 + 1] = accG + bg;
    out[ray * 3 + 2] = accB + bg;
  }
}

extern "C" void kernel_launch(void* const* d_in, const int* in_sizes, int n_in,
                              void* d_out, int out_size, void* d_ws, size_t ws_size,
                              hipStream_t stream) {
  const float* rays_o = (const float*)d_in[0];
  const float* rays_d = (const float*)d_in[1];
  const float* G1     = (const float*)d_in[2];
  const float* Fg     = (const float*)d_in[3];
  const float* s_w0   = (const float*)d_in[4];
  const float* s_w1   = (const float*)d_in[5];
  const float* s_w2   = (const float*)d_in[6];
  const float* c_w0   = (const float*)d_in[7];
  const float* c_w1   = (const float*)d_in[8];
  const float* c_w2   = (const float*)d_in[9];
  _Float16* wt = (_Float16*)d_ws;
  float* outp = (float*)d_out;

  int B = in_sizes[0] / 3;  // 8192 rays

  prep_weights<<<64, 256, 0, stream>>>(s_w0, s_w1, s_w2, c_w0, c_w1, c_w2, wt);
  nerf_main<<<B / 4, 128, 0, stream>>>(rays_o, rays_d, G1, Fg, wt, outp);
}
